// SelfAttentionBlock_90709709291587
// MI455X (gfx1250) — matrix operand order repaired and checked by reading, not compiled, not run on hardware
//
#include <hip/hip_runtime.h>

// Self-attention block for MI455X (gfx1250): bf16 WMMA GEMMs + flash attention
// with block-cooperative LDS staging of K/V tiles.
// B=2, H=W=64 -> L=4096, C=512, 8 heads x d=64, 32 groups x 16 ch.

typedef __attribute__((ext_vector_type(16))) __bf16 v16bf;
typedef __attribute__((ext_vector_type(8)))  float  v8f;

#define LSEQ  4096
#define CCH   512
#define NH    8
#define DH    64
#define NGRP  32

// ---------------------------------------------------------------------------
// WMMA helpers (CDNA5 16x16x32 bf16, f32 accum)
// ---------------------------------------------------------------------------
__device__ __forceinline__ v8f wmma_bf16(v16bf a, v16bf b, v8f c) {
  // (neg_a, A, neg_b, B, c_mod, C, reuse_a, reuse_b)
  return __builtin_amdgcn_wmma_f32_16x16x32_bf16(false, a, false, b, (short)0, c,
                                                 false, false);
}

// A-operand (16x32, MxK) per ISA 7.12.2:
//   lanes 0-15 : row m = lane,    K = {0..7, 16..23}
//   lanes 16-31: row m = lane-16, K = {8..15, 24..31}
__device__ __forceinline__ v16bf load_A_tile(const __bf16* p, int ld) {
  const int lane = threadIdx.x & 31;
  const int m  = lane & 15;
  const int kb = (lane < 16) ? 0 : 8;
  const __bf16* r = p + (size_t)m * ld + kb;
  union { v16bf v; uint4 q[2]; } u;
  u.q[0] = *reinterpret_cast<const uint4*>(r);        // K = kb .. kb+7
  u.q[1] = *reinterpret_cast<const uint4*>(r + 16);   // K = 16+kb .. 16+kb+7
  return u.v;
}

// B-operand (32x16, KxN) from N-major ("transposed") storage BT[n][k]:
//   lanes 0-15: col n=lane, K={0..7,16..23}; lanes 16-31: col n=lane-16, K={8..15,24..31}.
__device__ __forceinline__ v16bf load_B_tileT(const __bf16* pT, int ld) {
  const int lane = threadIdx.x & 31;
  const int n  = lane & 15;
  const int kb = (lane < 16) ? 0 : 8;
  const __bf16* r = pT + (size_t)n * ld + kb;
  union { v16bf v; uint4 q[2]; } u;
  u.q[0] = *reinterpret_cast<const uint4*>(r);
  u.q[1] = *reinterpret_cast<const uint4*>(r + 16);
  return u.v;
}

// ---------------------------------------------------------------------------
// Weight packing (f32 -> bf16, transposed so B-tiles are contiguous)
// ---------------------------------------------------------------------------
__global__ void pack_qkv_w(const float* __restrict__ Wq, const float* __restrict__ Wk,
                           const float* __restrict__ Wv, __bf16* __restrict__ wT) {
  const size_t i = (size_t)blockIdx.x * 256 + threadIdx.x;   // 1536*512 elements
  const int k = (int)(i & 511);
  const int n = (int)(i >> 9);       // 0..1535
  const int p = n >> 9;              // 0:q 1:k 2:v
  const int c = n & 511;
  const float* W = (p == 0) ? Wq : (p == 1) ? Wk : Wv;
  wT[i] = (__bf16)W[(size_t)k * CCH + c];
}

__global__ void pack_o_w(const float* __restrict__ Wo, __bf16* __restrict__ wT) {
  const size_t i = (size_t)blockIdx.x * 256 + threadIdx.x;   // 512*512 elements
  const int k = (int)(i & 511);
  const int n = (int)(i >> 9);
  wT[i] = (__bf16)Wo[(size_t)k * CCH + n];
}

// ---------------------------------------------------------------------------
// GroupNorm
// ---------------------------------------------------------------------------
__global__ void gn_stats(const float* __restrict__ x, float* __restrict__ mean,
                         float* __restrict__ rstd) {
  const int bg = blockIdx.x;          // b*32+g, 64 blocks
  const int b = bg >> 5, g = bg & 31;
  const float* base = x + (size_t)b * LSEQ * CCH + g * 16;
  float s = 0.f, ss = 0.f;
  for (int i = threadIdx.x; i < LSEQ * 16; i += 256) {
    const int l = i >> 4, c = i & 15;
    const float v = base[(size_t)l * CCH + c];
    s += v; ss += v * v;
  }
  __shared__ float sh[256], sh2[256];
  sh[threadIdx.x] = s; sh2[threadIdx.x] = ss;
  __syncthreads();
  for (int st = 128; st > 0; st >>= 1) {
    if (threadIdx.x < st) { sh[threadIdx.x] += sh[threadIdx.x + st];
                            sh2[threadIdx.x] += sh2[threadIdx.x + st]; }
    __syncthreads();
  }
  if (threadIdx.x == 0) {
    const float inv = 1.f / (float)(LSEQ * 16);
    const float m = sh[0] * inv;
    const float v = sh2[0] * inv - m * m;
    mean[bg] = m;
    rstd[bg] = rsqrtf(v + 1e-6f);
  }
}

__global__ void gn_apply(const float* __restrict__ x, const float* __restrict__ mean,
                         const float* __restrict__ rstd, const float* __restrict__ sc,
                         const float* __restrict__ bi, __bf16* __restrict__ xb) {
  const size_t i = (size_t)blockIdx.x * 256 + threadIdx.x;   // 8192*512
  const int c = (int)(i & 511);
  const size_t m = i >> 9;
  const int b = (int)(m >> 12);
  const int bg = b * NGRP + (c >> 4);
  const float v = (x[i] - mean[bg]) * rstd[bg] * sc[c] + bi[c];
  xb[i] = (__bf16)v;
}

// ---------------------------------------------------------------------------
// QKV projection: [8192,512] x [512,1536], wave = 16x64 tile
// ---------------------------------------------------------------------------
__global__ void qkv_gemm(const __bf16* __restrict__ xb, const __bf16* __restrict__ wT,
                         const float* __restrict__ bq, const float* __restrict__ bk,
                         const float* __restrict__ bv, __bf16* __restrict__ qg,
                         __bf16* __restrict__ kg, __bf16* __restrict__ vg) {
  const int wv = threadIdx.x >> 5, lane = threadIdx.x & 31;
  const int gw = blockIdx.x * 8 + wv;          // 12288 waves
  const int mt = gw / 24;
  const int ntile = gw % 24;
  const int n0 = ntile * 64;
  const __bf16* arow = xb + (size_t)mt * 16 * CCH;
  v8f acc[4] = {};
  for (int k0 = 0; k0 < CCH; k0 += 32) {
    const v16bf a = load_A_tile(arow + k0, CCH);
#pragma unroll
    for (int t = 0; t < 4; ++t) {
      const v16bf bt = load_B_tileT(wT + (size_t)(n0 + t * 16) * CCH + k0, CCH);
      acc[t] = wmma_bf16(a, bt, acc[t]);
    }
  }
  const int proj = ntile >> 3, head = ntile & 7;
  const int hi = (lane >= 16) ? 8 : 0, col = lane & 15;
#pragma unroll
  for (int t = 0; t < 4; ++t) {
    const int dd = t * 16 + col;
    const int c = head * DH + dd;
    const float bias = (proj == 0) ? bq[c] : (proj == 1) ? bk[c] : bv[c];
#pragma unroll
    for (int j = 0; j < 8; ++j) {
      const int m = mt * 16 + j + hi;
      const int b = m >> 12, l = m & (LSEQ - 1);
      const float val = acc[t][j] + bias;
      const size_t bh = (size_t)(b * NH + head);
      if (proj == 0)      qg[(bh * LSEQ + l) * DH + dd] = (__bf16)(val * 0.125f); // fold 1/sqrt(d)
      else if (proj == 1) kg[(bh * LSEQ + l) * DH + dd] = (__bf16)val;
      else                vg[(bh * DH + dd) * LSEQ + l] = (__bf16)val;            // V transposed
    }
  }
}

// ---------------------------------------------------------------------------
// Flash attention. Block = 8 waves = 8 query tiles of the SAME (b,h).
// Each 32-key step: the block cooperatively stages K (32x64) and V^T (64x32)
// into LDS once; all 8 waves consume them via ds_load_b128 -> 8x less L2
// traffic, attention becomes WMMA-bound.
// ---------------------------------------------------------------------------
__global__ void attn_kernel(const __bf16* __restrict__ qg, const __bf16* __restrict__ kg,
                            const __bf16* __restrict__ vg, __bf16* __restrict__ og) {
  __shared__ __bf16 kt[32][64];        // [key][d]      4 KB
  __shared__ __bf16 vt[64][32];        // [d][key]      4 KB
  __shared__ __bf16 plds[8][16][32];   // per-wave P    8 KB
  const int tid = threadIdx.x;
  const int wv = tid >> 5;
  const int lane = tid & 31;
  const int bh = blockIdx.x >> 5;              // 512 blocks, 32 per (b,h)
  const int mt = (blockIdx.x & 31) * 8 + wv;   // query tile 0..255
  const int m0 = mt * 16;
  const int b = bh >> 3, h = bh & 7;

  const __bf16* qp = qg + (size_t)bh * LSEQ * DH;
  const __bf16* kp = kg + (size_t)bh * LSEQ * DH;
  const __bf16* vp = vg + (size_t)bh * DH * LSEQ;

  const v16bf qa0 = load_A_tile(qp + (size_t)m0 * DH, DH);       // d cols 0..31
  const v16bf qa1 = load_A_tile(qp + (size_t)m0 * DH + 32, DH);  // d cols 32..63

  // staging addresses for this thread
  const int skey = tid >> 3, soff = (tid & 7) * 8;   // K: 32 rows x 8 uint4
  const int sdd  = tid >> 2, soff2 = (tid & 3) * 8;  // V: 64 rows x 4 uint4

  float mrow[8], lrow[8];
  v8f o[4] = {};
#pragma unroll
  for (int j = 0; j < 8; ++j) { mrow[j] = -1e30f; lrow[j] = 0.f; }

  __bf16 (*pl)[32] = plds[wv];
  const int hi = (lane >= 16) ? 8 : 0;
  const int col = lane & 15;

  for (int j0 = 0; j0 < LSEQ; j0 += 32) {
    __syncthreads();   // previous iteration's LDS reads done before overwrite
    // cooperative staging: K tile and transposed V tile (one b128 each)
    *reinterpret_cast<uint4*>(&kt[skey][soff]) =
        *reinterpret_cast<const uint4*>(kp + (size_t)(j0 + skey) * DH + soff);
    *reinterpret_cast<uint4*>(&vt[sdd][soff2]) =
        *reinterpret_cast<const uint4*>(vp + (size_t)sdd * LSEQ + j0 + soff2);
    // prefetch next tiles into cache while we compute
    const int jn = (j0 + 32) & (LSEQ - 1);
    __builtin_prefetch(kp + (size_t)(jn + skey) * DH + soff, 0, 1);
    __builtin_prefetch(vp + (size_t)sdd * LSEQ + jn + soff2, 0, 1);
    __syncthreads();

    // S = Q K^T for 32 keys (two 16-wide N tiles, K-dim = d in two 32-steps)
    v8f s0 = {}, s1 = {};
#pragma unroll
    for (int s = 0; s < 2; ++s) {
      const v16bf a = s ? qa1 : qa0;
      const v16bf b0 = load_B_tileT(&kt[0][0]  + 32 * s, DH);
      const v16bf b1 = load_B_tileT(&kt[16][0] + 32 * s, DH);
      s0 = wmma_bf16(a, b0, s0);
      s1 = wmma_bf16(a, b1, s1);
    }
    // online softmax (row = j + hi; 16-lane-half reductions match D layout)
    float fs[8];
#pragma unroll
    for (int j = 0; j < 8; ++j) {
      float t = fmaxf(s0[j], s1[j]);
      t = fmaxf(t, __shfl_xor(t, 1));
      t = fmaxf(t, __shfl_xor(t, 2));
      t = fmaxf(t, __shfl_xor(t, 4));
      t = fmaxf(t, __shfl_xor(t, 8));
      const float mn = fmaxf(mrow[j], t);
      fs[j] = __expf(mrow[j] - mn);
      mrow[j] = mn;
      const __bf16 p0 = (__bf16)__expf(s0[j] - mn);
      const __bf16 p1 = (__bf16)__expf(s1[j] - mn);
      pl[j + hi][col]      = p0;
      pl[j + hi][col + 16] = p1;
      // denominator from the same bf16-rounded P the PV WMMA consumes
      float rs = (float)p0 + (float)p1;
      rs += __shfl_xor(rs, 1);
      rs += __shfl_xor(rs, 2);
      rs += __shfl_xor(rs, 4);
      rs += __shfl_xor(rs, 8);
      lrow[j] = lrow[j] * fs[j] + rs;
    }
#pragma unroll
    for (int t = 0; t < 4; ++t)
#pragma unroll
      for (int j = 0; j < 8; ++j) o[t][j] *= fs[j];
    // P back in A-operand layout (same-wave LDS is in-order)
    const v16bf pa = load_A_tile(&pl[0][0], 32);
#pragma unroll
    for (int t = 0; t < 4; ++t) {
      const v16bf vb = load_B_tileT(&vt[t * 16][0], 32);
      o[t] = wmma_bf16(pa, vb, o[t]);
    }
  }
#pragma unroll
  for (int t = 0; t < 4; ++t) {
    const int dd = t * 16 + col;
#pragma unroll
    for (int j = 0; j < 8; ++j) {
      const int l = m0 + j + hi;
      const float val = o[t][j] / lrow[j];
      og[((size_t)b * LSEQ + l) * CCH + h * DH + dd] = (__bf16)val;
    }
  }
}

// ---------------------------------------------------------------------------
// Output projection + bias + residual (f32 out)
// ---------------------------------------------------------------------------
__global__ void out_gemm(const __bf16* __restrict__ ob, const __bf16* __restrict__ wT,
                         const float* __restrict__ bo, const float* __restrict__ resid,
                         float* __restrict__ out) {
  const int wv = threadIdx.x >> 5, lane = threadIdx.x & 31;
  const int gw = blockIdx.x * 8 + wv;          // 4096 waves
  const int mt = gw >> 3, nt = gw & 7, n0 = nt * 64;
  const __bf16* arow = ob + (size_t)mt * 16 * CCH;
  v8f acc[4] = {};
  for (int k0 = 0; k0 < CCH; k0 += 32) {
    const v16bf a = load_A_tile(arow + k0, CCH);
#pragma unroll
    for (int t = 0; t < 4; ++t) {
      const v16bf bt = load_B_tileT(wT + (size_t)(n0 + t * 16) * CCH + k0, CCH);
      acc[t] = wmma_bf16(a, bt, acc[t]);
    }
  }
  const int hi = (lane >= 16) ? 8 : 0, col = lane & 15;
#pragma unroll
  for (int t = 0; t < 4; ++t) {
    const int n = n0 + t * 16 + col;
    const float bias = bo[n];
#pragma unroll
    for (int j = 0; j < 8; ++j) {
      const int m = mt * 16 + j + hi;
      const size_t idx = (size_t)m * CCH + n;
      out[idx] = acc[t][j] + bias + resid[idx];
    }
  }
}

// ---------------------------------------------------------------------------
// Launch
// ---------------------------------------------------------------------------
extern "C" void kernel_launch(void* const* d_in, const int* in_sizes, int n_in,
                              void* d_out, int out_size, void* d_ws, size_t ws_size,
                              hipStream_t stream) {
  (void)in_sizes; (void)n_in; (void)out_size; (void)ws_size;
  const float* x   = (const float*)d_in[0];
  const float* gsc = (const float*)d_in[1];
  const float* gbi = (const float*)d_in[2];
  const float* Wq  = (const float*)d_in[3];
  const float* bq  = (const float*)d_in[4];
  const float* Wk  = (const float*)d_in[5];
  const float* bk  = (const float*)d_in[6];
  const float* Wv  = (const float*)d_in[7];
  const float* bv  = (const float*)d_in[8];
  const float* Wo  = (const float*)d_in[9];
  const float* bo  = (const float*)d_in[10];
  float* out = (float*)d_out;

  // workspace carve (~45 MB)
  char* p = (char*)d_ws;
  float*  mean  = (float*)p;                              // 64 f32
  float*  rstd  = (float*)(p + 256);                      // 64 f32
  __bf16* xb    = (__bf16*)(p + 1024);                    // [8192][512]
  __bf16* wqkvT = xb    + (size_t)8192 * 512;             // [1536][512]
  __bf16* woT   = wqkvT + (size_t)1536 * 512;             // [512][512]
  __bf16* qg    = woT   + (size_t)512 * 512;              // [16][4096][64]
  __bf16* kg    = qg    + (size_t)16 * 4096 * 64;         // [16][4096][64]
  __bf16* vg    = kg    + (size_t)16 * 4096 * 64;         // [16][64][4096]
  __bf16* og    = vg    + (size_t)16 * 4096 * 64;         // [8192][512]

  pack_qkv_w <<<3072, 256, 0, stream>>>(Wq, Wk, Wv, wqkvT);
  pack_o_w   <<<1024, 256, 0, stream>>>(Wo, woT);
  gn_stats   <<<64,   256, 0, stream>>>(x, mean, rstd);
  gn_apply   <<<16384,256, 0, stream>>>(x, mean, rstd, gsc, gbi, xb);
  qkv_gemm   <<<1536, 256, 0, stream>>>(xb, wqkvT, bq, bk, bv, qg, kg, vg);
  attn_kernel<<<512,  256, 0, stream>>>(qg, kg, vg, og);
  out_gemm   <<<512,  256, 0, stream>>>(og, woT, bo, x, out);
}
